// HMVPCollisionDetector_7868380086802
// MI455X (gfx1250) — matrix operand order repaired and checked by reading, not compiled, not run on hardware
//
#include <hip/hip_runtime.h>
#include <hip/hip_bf16.h>
#include <math.h>

typedef __attribute__((ext_vector_type(16))) _Float16 v16h;
typedef __attribute__((ext_vector_type(8)))  _Float16 v8h;
typedef __attribute__((ext_vector_type(8)))  float    v8f;

#define NIMG 192          // B(32) * V(6)
#define GN_EPS 1e-5f

// ---------------------------------------------------------------------------
// Fused depthwise 5x5 Gaussian blur (zero pad) + 2x2 avg pool, stride 2.
// ---------------------------------------------------------------------------
__global__ void k_downsample(const float* __restrict__ in, float* __restrict__ out,
                             int H, int W, float sigma) {
    int H2 = H >> 1, W2 = W >> 1;
    int total = NIMG * 2 * H2 * W2;
    int i = blockIdx.x * blockDim.x + threadIdx.x;
    if (i >= total) return;
    int x = i % W2; int t = i / W2;
    int y = t % H2; t /= H2;
    int c = t % 2;  int img = t / 2;

    float g[5]; float s = 0.f;
#pragma unroll
    for (int j = 0; j < 5; ++j) {
        float d = (float)j - 2.0f;
        g[j] = __expf(-(d * d) / (2.f * sigma * sigma));
        s += g[j];
    }
#pragma unroll
    for (int j = 0; j < 5; ++j) g[j] /= s;
    float wv[6];
#pragma unroll
    for (int tt = 0; tt < 6; ++tt) {
        float a = (tt < 5) ? g[tt] : 0.f;
        float b = (tt >= 1) ? g[tt - 1] : 0.f;
        wv[tt] = a + b;
    }

    const float* src = in + (size_t)(img * 2 + c) * H * W;
    float acc = 0.f;
    int y0 = 2 * y - 2, x0 = 2 * x - 2;
#pragma unroll
    for (int ty = 0; ty < 6; ++ty) {
        int iy = y0 + ty;
        if (iy < 0 || iy >= H) continue;
        float wy = wv[ty];
#pragma unroll
        for (int tx = 0; tx < 6; ++tx) {
            int ix = x0 + tx;
            if (ix < 0 || ix >= W) continue;
            acc += wy * wv[tx] * src[iy * W + ix];
        }
    }
    out[(size_t)(img * 2 + c) * H2 * W2 + y * W2 + x] = 0.25f * acc;
}

// ---------------------------------------------------------------------------
// conv1: 3x3, Cin=4, Cout=C, zero pad, + bias.  Output channel-last fp16.
// ---------------------------------------------------------------------------
template <int C, int H>
__global__ void k_conv1(const float* __restrict__ obj, const float* __restrict__ scn,
                        const float* __restrict__ w1, const float* __restrict__ b1,
                        _Float16* __restrict__ h1) {
    constexpr int W = H;
    constexpr int HW = H * W;
    __shared__ float sw[C * 36];
    for (int i = threadIdx.x; i < C * 36; i += blockDim.x) sw[i] = w1[i];
    __syncthreads();

    int idx = blockIdx.x * blockDim.x + threadIdx.x;
    if (idx >= NIMG * HW) return;
    int p = idx % HW; int img = idx / HW;
    int y = p / W, x = p % W;

    float patch[36];
#pragma unroll
    for (int ci = 0; ci < 4; ++ci) {
        const float* src = (ci < 2 ? obj : scn) + ((size_t)img * 2 + (ci & 1)) * HW;
#pragma unroll
        for (int dy = 0; dy < 3; ++dy)
#pragma unroll
            for (int dx = 0; dx < 3; ++dx) {
                int iy = y + dy - 1, ix = x + dx - 1;
                float v = (iy >= 0 && iy < H && ix >= 0 && ix < W) ? src[iy * W + ix] : 0.f;
                patch[ci * 9 + dy * 3 + dx] = v;
            }
    }
    _Float16* dst = h1 + (size_t)idx * C;
    for (int co = 0; co < C; ++co) {
        float acc = b1[co];
        const float* wrow = &sw[co * 36];
#pragma unroll
        for (int k = 0; k < 36; ++k) acc += wrow[k] * patch[k];
        dst[co] = (_Float16)acc;
    }
}

// ---------------------------------------------------------------------------
// GroupNorm: per-(img,group) sum/sumsq, one block each, LDS tree reduce.
// ---------------------------------------------------------------------------
template <int C, int CG>
__global__ void k_gn_stats(const _Float16* __restrict__ xbuf, float* __restrict__ stats,
                           int HW, int groups) {
    int blk = blockIdx.x;
    int grp = blk % groups; int img = blk / groups;
    const _Float16* src = xbuf + (size_t)img * HW * C;
    int n = HW * CG;
    float s = 0.f, ss = 0.f;
    for (int i = threadIdx.x; i < n; i += blockDim.x) {
        int p = i / CG; int cc = grp * CG + (i % CG);
        float v = (float)src[p * C + cc];
        s += v; ss += v * v;
    }
    __shared__ float rs[256], rss[256];
    rs[threadIdx.x] = s; rss[threadIdx.x] = ss;
    __syncthreads();
    for (int off = 128; off > 0; off >>= 1) {
        if ((int)threadIdx.x < off) {
            rs[threadIdx.x] += rs[threadIdx.x + off];
            rss[threadIdx.x] += rss[threadIdx.x + off];
        }
        __syncthreads();
    }
    if (threadIdx.x == 0) { stats[blk * 2] = rs[0]; stats[blk * 2 + 1] = rss[0]; }
}

__global__ void k_gn_coef(const float* __restrict__ stats, const float* __restrict__ gamma,
                          const float* __restrict__ beta, float* __restrict__ abuf,
                          float* __restrict__ bbuf, int C, int cg, int HW, int groups) {
    int i = blockIdx.x * blockDim.x + threadIdx.x;
    if (i >= NIMG * C) return;
    int c = i % C; int img = i / C;
    int g = c / cg;
    float n = (float)(cg * HW);
    float s = stats[(img * groups + g) * 2];
    float ss = stats[(img * groups + g) * 2 + 1];
    float mu = s / n;
    float var = ss / n - mu * mu;
    float rinv = rsqrtf(var + GN_EPS);
    float a = gamma[c] * rinv;
    abuf[i] = a;
    bbuf[i] = beta[c] - mu * a;
}

template <int C>
__global__ void k_gn_apply(_Float16* __restrict__ buf, const float* __restrict__ abuf,
                           const float* __restrict__ bbuf, int HW) {
    int i = blockIdx.x * blockDim.x + threadIdx.x;
    if (i >= NIMG * HW * C) return;
    int c = i % C; int img = i / (HW * C);
    float v = (float)buf[i];
    v = abuf[img * C + c] * v + bbuf[img * C + c];
    buf[i] = (_Float16)(v > 0.f ? v : 0.f);
}

// ---------------------------------------------------------------------------
// Pack w2 (C,C,3,3 fp32) into per-lane WMMA A-fragment order (fp16).
// K layout: k = tap*C + ci.  Contiguous A order (matches B loads):
// lane<16 covers k_local {0..15}; lane>=16 covers {16..31}.
// ---------------------------------------------------------------------------
template <int C>
__global__ void k_pack_w2(const float* __restrict__ w2, _Float16* __restrict__ aw) {
    constexpr int MT = C / 16;
    constexpr int KTOT = 9 * C;
    constexpr int KSTEPS = (KTOT + 31) / 32;
    constexpr int TOTAL = KSTEPS * MT * 32 * 16;
    int i = blockIdx.x * blockDim.x + threadIdx.x;
    if (i >= TOTAL) return;
    int e = i & 15;
    int lane = (i >> 4) & 31;
    int mt = (i >> 9) % MT;
    int kstep = i / (512 * MT);
    int base = (lane >= 16) ? 16 : 0;
    int klocal = base + e;
    int k = kstep * 32 + klocal;
    int m = mt * 16 + (lane & 15);
    float v = 0.f;
    if (k < KTOT) {
        int tap = k / C, ci = k % C;
        v = w2[(m * C + ci) * 9 + tap];
    }
    aw[i] = (_Float16)v;
}

// ---------------------------------------------------------------------------
// conv2 as implicit GEMM on v_wmma_f32_16x16x32_f16, fully unrolled k-loop.
// Grid: (HW/16, NIMG).  Block: one wave per 16-output-channel tile.
// ---------------------------------------------------------------------------
template <int C, int H>
__global__ void k_conv2_wmma(const _Float16* __restrict__ h1, const _Float16* __restrict__ aw,
                             const float* __restrict__ b2, _Float16* __restrict__ h2) {
    constexpr int W = H;
    constexpr int HW = H * W;
    constexpr int MT = C / 16;
    constexpr int KTOT = 9 * C;
    constexpr int KSTEPS = (KTOT + 31) / 32;

    int lane = threadIdx.x & 31;
    int wave = threadIdx.x >> 5;
    int img = blockIdx.y;
    int tile = blockIdx.x;
    int n = lane & 15;
    int s = tile * 16 + n;
    int h = s / W, w = s % W;          // W is a power-of-two constant -> shifts
    int hi = (lane >= 16) ? 1 : 0;
    const _Float16* src = h1 + (size_t)img * HW * C;
    const _Float16* abase = aw + ((size_t)wave * 32 + lane) * 16;

    v8f acc;
#pragma unroll
    for (int r = 0; r < 8; ++r) acc[r] = 0.f;

#pragma unroll
    for (int kstep = 0; kstep < KSTEPS; ++kstep) {
        const v16h a = *(const v16h*)(abase + (size_t)kstep * MT * 512);
        if (kstep + 1 < KSTEPS)
            __builtin_prefetch(abase + (size_t)(kstep + 1) * MT * 512, 0, 1);

        int kst = kstep * 32 + (hi ? 16 : 0);
        v16h b;
#pragma unroll
        for (int e = 0; e < 16; ++e) b[e] = (_Float16)0;
        if (kst < KTOT) {
            int tap = kst / C, ci0 = kst % C;   // constant-folded per unrolled step
            int dy = tap / 3 - 1, dx = tap % 3 - 1;
            int iy = h + dy, ix = w + dx;
            if (iy >= 0 && iy < H && ix >= 0 && ix < W)
                b = *(const v16h*)(src + (size_t)(iy * W + ix) * C + ci0);
        }
        acc = __builtin_amdgcn_wmma_f32_16x16x32_f16(false, a, false, b,
                                                     (short)0, acc, false, false);
    }

    int m0 = wave * 16 + (hi ? 8 : 0);
    v8h outv;
#pragma unroll
    for (int r = 0; r < 8; ++r) outv[r] = (_Float16)(acc[r] + b2[m0 + r]);
    *(v8h*)(h2 + ((size_t)img * HW + s) * C + m0) = outv;
}

// ---------------------------------------------------------------------------
// conv1x1 (C->2) + spatial mean: one block per image, LDS reduce.
// ---------------------------------------------------------------------------
template <int C>
__global__ void k_conv3(const _Float16* __restrict__ h2, const float* __restrict__ w3,
                        const float* __restrict__ b3, float* __restrict__ logits,
                        int HW) {
    int img = blockIdx.x;
    __shared__ float sw3[2 * C];
    for (int i = threadIdx.x; i < 2 * C; i += blockDim.x) sw3[i] = w3[i];
    __syncthreads();
    const _Float16* src = h2 + (size_t)img * HW * C;
    float p0 = 0.f, p1 = 0.f;
    for (int p = threadIdx.x; p < HW; p += blockDim.x) {
        const _Float16* row = src + p * C;
#pragma unroll
        for (int c = 0; c < C; ++c) {
            float v = (float)row[c];
            p0 += sw3[c] * v;
            p1 += sw3[C + c] * v;
        }
    }
    __shared__ float r0[256], r1[256];
    r0[threadIdx.x] = p0; r1[threadIdx.x] = p1;
    __syncthreads();
    for (int off = 128; off > 0; off >>= 1) {
        if ((int)threadIdx.x < off) {
            r0[threadIdx.x] += r0[threadIdx.x + off];
            r1[threadIdx.x] += r1[threadIdx.x + off];
        }
        __syncthreads();
    }
    if (threadIdx.x == 0) {
        float inv = 1.f / (float)HW;
        logits[img * 2 + 0] = r0[0] * inv + b3[0];
        logits[img * 2 + 1] = r1[0] * inv + b3[1];
    }
}

// ---------------------------------------------------------------------------
// Early-out state machine over levels; one thread per batch element.
// ---------------------------------------------------------------------------
__global__ void k_finalize(const float* __restrict__ logits, float* __restrict__ out) {
    int b = threadIdx.x;
    if (b >= 32) return;
    float active = 1.f, cost = 0.f, coll = 0.f;
    for (int level = 0; level < 4; ++level) {
        const float* L = logits + level * NIMG * 2;
        float l0 = 0.f, l1 = 0.f;
#pragma unroll
        for (int v = 0; v < 6; ++v) {
            l0 += L[(b * 6 + v) * 2 + 0];
            l1 += L[(b * 6 + v) * 2 + 1];
        }
        l0 *= (1.f / 6.f); l1 *= (1.f / 6.f);
        float c  = active / (1.f + __expf(-l0));
        float sd = active / (1.f + __expf(-l1));
        coll = c;
        cost += active * (float)(level + 1);
        if (level < 3) {
            bool amb  = (c > 0.1f) && (c < 0.9f);
            bool cont = (sd > 0.5f) || amb;
            active *= cont ? 1.f : 0.f;
        }
    }
    out[b]      = coll;
    out[32 + b] = cost;
}

// ---------------------------------------------------------------------------
// Per-level driver, templated so all divisions/loops constant-fold.
// ---------------------------------------------------------------------------
template <int LEVEL>
static void run_level(void* const* d_in, const float* obj, const float* scn,
                      const float* in_o, const float* in_s,
                      _Float16* h1, _Float16* h2, _Float16* aw,
                      float* stats, float* abuf, float* bbuf, float* logits,
                      hipStream_t stream) {
    constexpr int C = 16 * (LEVEL + 1);
    constexpr int H = 8 << LEVEL;
    constexpr int HW = H * H;
    constexpr int GROUPS = (C / 4 < 8) ? (C / 4) : 8;
    constexpr int CG = C / GROUPS;
    constexpr int KSTEPS = (9 * C + 31) / 32;
    (void)obj; (void)scn;

    int pb = 2 + 10 * LEVEL;
    const float* w1  = (const float*)d_in[pb + 0];
    const float* b1  = (const float*)d_in[pb + 1];
    const float* g1  = (const float*)d_in[pb + 2];
    const float* bt1 = (const float*)d_in[pb + 3];
    const float* w2  = (const float*)d_in[pb + 4];
    const float* b2  = (const float*)d_in[pb + 5];
    const float* gm2 = (const float*)d_in[pb + 6];
    const float* bt2 = (const float*)d_in[pb + 7];
    const float* w3  = (const float*)d_in[pb + 8];
    const float* b3  = (const float*)d_in[pb + 9];

    int t1 = NIMG * HW;
    k_conv1<C, H><<<(t1 + 255) / 256, 256, 0, stream>>>(in_o, in_s, w1, b1, h1);
    k_gn_stats<C, CG><<<NIMG * GROUPS, 256, 0, stream>>>(h1, stats, HW, GROUPS);
    k_gn_coef<<<(NIMG * C + 255) / 256, 256, 0, stream>>>(stats, g1, bt1, abuf, bbuf, C, CG, HW, GROUPS);
    k_gn_apply<C><<<(NIMG * HW * C + 255) / 256, 256, 0, stream>>>(h1, abuf, bbuf, HW);

    int tp = KSTEPS * (C / 16) * 512;
    k_pack_w2<C><<<(tp + 255) / 256, 256, 0, stream>>>(w2, aw);
    dim3 gconv(HW / 16, NIMG);
    k_conv2_wmma<C, H><<<gconv, 32 * (C / 16), 0, stream>>>(h1, aw, b2, h2);

    k_gn_stats<C, CG><<<NIMG * GROUPS, 256, 0, stream>>>(h2, stats, HW, GROUPS);
    k_gn_coef<<<(NIMG * C + 255) / 256, 256, 0, stream>>>(stats, gm2, bt2, abuf, bbuf, C, CG, HW, GROUPS);
    k_gn_apply<C><<<(NIMG * HW * C + 255) / 256, 256, 0, stream>>>(h2, abuf, bbuf, HW);

    k_conv3<C><<<NIMG, 256, 0, stream>>>(h2, w3, b3, logits + LEVEL * NIMG * 2, HW);
}

// ---------------------------------------------------------------------------
extern "C" void kernel_launch(void* const* d_in, const int* in_sizes, int n_in,
                              void* d_out, int out_size, void* d_ws, size_t ws_size,
                              hipStream_t stream) {
    (void)in_sizes; (void)n_in; (void)out_size; (void)ws_size;
    const float* obj = (const float*)d_in[0];
    const float* scn = (const float*)d_in[1];

    char* ws = (char*)d_ws;
    size_t off = 0;
    auto alloc = [&](size_t bytes) -> char* {
        char* p = ws + off;
        off += (bytes + 255) & ~(size_t)255;
        return p;
    };

    const int Hs[4] = {8, 16, 32, 64};
    float* pobj[3]; float* pscn[3];
    for (int l = 0; l < 3; ++l) pobj[l] = (float*)alloc((size_t)NIMG * 2 * Hs[l] * Hs[l] * 4);
    for (int l = 0; l < 3; ++l) pscn[l] = (float*)alloc((size_t)NIMG * 2 * Hs[l] * Hs[l] * 4);
    _Float16* h1 = (_Float16*)alloc((size_t)NIMG * 4096 * 64 * 2);
    _Float16* h2 = (_Float16*)alloc((size_t)NIMG * 4096 * 64 * 2);
    _Float16* aw = (_Float16*)alloc(1 << 17);
    float* stats  = (float*)alloc((size_t)NIMG * 8 * 2 * 4);
    float* abuf   = (float*)alloc((size_t)NIMG * 64 * 4);
    float* bbuf   = (float*)alloc((size_t)NIMG * 64 * 4);
    float* logits = (float*)alloc((size_t)4 * NIMG * 2 * 4);

    // pyramids: 64 -> 32 (sigma 3) -> 16 (sigma 2) -> 8 (sigma 1)
    {
        int n;
        n = NIMG * 2 * 32 * 32;
        k_downsample<<<(n + 255) / 256, 256, 0, stream>>>(obj, pobj[2], 64, 64, 3.f);
        k_downsample<<<(n + 255) / 256, 256, 0, stream>>>(scn, pscn[2], 64, 64, 3.f);
        n = NIMG * 2 * 16 * 16;
        k_downsample<<<(n + 255) / 256, 256, 0, stream>>>(pobj[2], pobj[1], 32, 32, 2.f);
        k_downsample<<<(n + 255) / 256, 256, 0, stream>>>(pscn[2], pscn[1], 32, 32, 2.f);
        n = NIMG * 2 * 8 * 8;
        k_downsample<<<(n + 255) / 256, 256, 0, stream>>>(pobj[1], pobj[0], 16, 16, 1.f);
        k_downsample<<<(n + 255) / 256, 256, 0, stream>>>(pscn[1], pscn[0], 16, 16, 1.f);
    }

    run_level<0>(d_in, obj, scn, pobj[0], pscn[0], h1, h2, aw, stats, abuf, bbuf, logits, stream);
    run_level<1>(d_in, obj, scn, pobj[1], pscn[1], h1, h2, aw, stats, abuf, bbuf, logits, stream);
    run_level<2>(d_in, obj, scn, pobj[2], pscn[2], h1, h2, aw, stats, abuf, bbuf, logits, stream);
    run_level<3>(d_in, obj, scn, obj, scn, h1, h2, aw, stats, abuf, bbuf, logits, stream);

    k_finalize<<<1, 32, 0, stream>>>(logits, (float*)d_out);
}
